// EngramNew_2070174237244
// MI455X (gfx1250) — matrix operand order repaired and checked by reading, not compiled, not run on hardware
//
#include <hip/hip_runtime.h>
#include <hip/hip_bf16.h>
#include <math.h>
#include <stdint.h>

// ---------------------------------------------------------------------------
// Problem dimensions (fixed by the reference: B=2, T=4096, G=4, C=1024, E=1024)
// ---------------------------------------------------------------------------
static constexpr int Bb = 2;
static constexpr int Tt = 4096;
static constexpr int Gg = 4;
static constexpr int Cc = 1024;
static constexpr int Ee = 1024;
static constexpr int Mm = Bb * Tt;   // 8192 GEMM rows
static constexpr int Nk = Gg * Cc;   // 4096 key-proj cols
static constexpr int Nv = Cc;        // 1024 value-proj cols

typedef __attribute__((ext_vector_type(16))) __bf16 v16bf;
typedef __attribute__((ext_vector_type(8)))  __bf16 v8bf;
typedef __attribute__((ext_vector_type(8)))  float  v8f;

// ---------------------------------------------------------------------------
// CDNA5 async copy: global -> LDS with no VGPR round trip, tracked by ASYNCcnt.
// VDST VGPR holds the per-lane LDS byte address (low 32 bits of the generic
// shared-aperture pointer), VADDR the 64-bit global address.
// ---------------------------------------------------------------------------
__device__ __forceinline__ void async_copy_b128(unsigned ldsOff, const void* gptr)
{
    unsigned long long ga = (unsigned long long)(uintptr_t)gptr;
    asm volatile("global_load_async_to_lds_b128 %0, %1, off"
                 :: "v"(ldsOff), "v"(ga) : "memory");
}

__device__ __forceinline__ void wait_async0()
{
    asm volatile("s_wait_asynccnt 0x0" ::: "memory");
}

// ---------------------------------------------------------------------------
// Kernel 1: split fp32 -> bf16 hi/lo pair (bf16x3 emulation of fp32 GEMM)
// ---------------------------------------------------------------------------
__global__ void __launch_bounds__(256)
split_bf16_kernel(const float* __restrict__ x,
                  __bf16* __restrict__ hi, __bf16* __restrict__ lo, long n)
{
    long stride = (long)gridDim.x * blockDim.x;
    for (long i = (long)blockIdx.x * blockDim.x + threadIdx.x; i < n; i += stride) {
        float v = x[i];
        __bf16 h = (__bf16)v;          // RNE round to bf16
        float hf = (float)h;
        hi[i] = h;
        lo[i] = (__bf16)(v - hf);      // residual mantissa bits
    }
}

// ---------------------------------------------------------------------------
// Kernel 2: GEMM out[m,n] = sum_k A[m,k]*W[n,k] + bias[n], bf16x3 via WMMA.
// Block = 8 waves = 128(M) x 64(N); wave = 16(M) x 64(N) strip (4 v8f accum).
// B tile (64 rows x 32 k, hi+lo = 8KB) is shared by all 8 waves -> staged once
// per k-step into LDS via global_load_async_to_lds_b128 (double buffered,
// padded stride 40 elems = 80B to spread b128 reads across bank groups).
// A lane layout (16-bit A 16x32):  lane l holds row M = l&15,
//   element j<8 -> K = (l>>4)*8 + j ; j>=8 -> K = 16 + (l>>4)*8 + (j-8).
// B lane layout (16-bit B 32x16):  lane l holds col N = l&15,
//   element j<8 -> K = (l>>4)*8 + j ; j>=8 -> K = 16 + (l>>4)*8 + (j-8).
// ---------------------------------------------------------------------------
__global__ void __launch_bounds__(256)
gemm_bf16x3_wmma(const __bf16* __restrict__ Ahi, const __bf16* __restrict__ Alo,
                 const __bf16* __restrict__ Whi, const __bf16* __restrict__ Wlo,
                 const float* __restrict__ bias, float* __restrict__ out,
                 int N, int Kd)
{
    constexpr int RS = 40;                       // padded LDS row stride (elems)
    __shared__ alignas(32) __bf16 Bsh[2][2][64 * RS];   // 20 KB

    const int tid  = threadIdx.x;
    const int lane = tid & 31;
    const int wave = tid >> 5;
    const int lm   = lane & 15;
    const int kg   = lane >> 4;                  // half-wave selector
    const int mb   = blockIdx.y * 128 + wave * 16;
    const int nb   = blockIdx.x * 64;

    // --- cooperative staging role: thread -> (row 0..63, k-quarter 0..3) ---
    const int srow = tid >> 2;
    const int skq  = tid & 3;
    const __bf16* wHiS = Whi + (size_t)(nb + srow) * Kd + skq * 8;
    const __bf16* wLoS = Wlo + (size_t)(nb + srow) * Kd + skq * 8;
    const int sdst = srow * RS + skq * 8;

    auto stage = [&](int kb, int buf) {
        async_copy_b128((unsigned)(uintptr_t)&Bsh[buf][0][sdst], wHiS + kb);
        async_copy_b128((unsigned)(uintptr_t)&Bsh[buf][1][sdst], wLoS + kb);
    };

    // --- per-lane A pointers (streamed from global; hits L2 across N-blocks) --
    const __bf16* aHiRow = Ahi + (size_t)(mb + lm) * Kd + kg * 8;
    const __bf16* aLoRow = Alo + (size_t)(mb + lm) * Kd + kg * 8;

    auto loadB = [&](int buf, int h, int t) -> v16bf {
        const __bf16* p = &Bsh[buf][h][(t * 16 + lm) * RS + kg * 8];
        v8bf b0 = *(const v8bf*)p;               // ds_load_b128
        v8bf b1 = *(const v8bf*)(p + 16);
        v16bf r;
#pragma unroll
        for (int i = 0; i < 8; ++i) { r[i] = b0[i]; r[i + 8] = b1[i]; }
        return r;
    };

    v8f acc[4] = {v8f{}, v8f{}, v8f{}, v8f{}};

    stage(0, 0);
    wait_async0();
    __syncthreads();

    for (int kb = 0; kb < Kd; kb += 32) {
        const int cur = (kb >> 5) & 1;
        if (kb + 32 < Kd) stage(kb + 32, cur ^ 1);   // uniform branch

        // prefetch A four k-steps ahead (global_prefetch_b8)
        if (kb + 128 < Kd) {
            __builtin_prefetch(aHiRow + kb + 128, 0, 1);
            __builtin_prefetch(aLoRow + kb + 128, 0, 1);
        }

        // A tile: two contiguous 8-element runs per lane, for hi and lo
        v8bf ah0 = *(const v8bf*)(aHiRow + kb);
        v8bf ah1 = *(const v8bf*)(aHiRow + kb + 16);
        v8bf al0 = *(const v8bf*)(aLoRow + kb);
        v8bf al1 = *(const v8bf*)(aLoRow + kb + 16);
        v16bf ahi, alo;
#pragma unroll
        for (int i = 0; i < 8; ++i) {
            ahi[i] = ah0[i]; ahi[i + 8] = ah1[i];
            alo[i] = al0[i]; alo[i + 8] = al1[i];
        }

#pragma unroll
        for (int t = 0; t < 4; ++t) {
            v16bf bh = loadB(cur, 0, t);
            v16bf bl = loadB(cur, 1, t);
            acc[t] = __builtin_amdgcn_wmma_f32_16x16x32_bf16(
                false, ahi, false, bh, (short)0, acc[t], false, false);
            acc[t] = __builtin_amdgcn_wmma_f32_16x16x32_bf16(
                false, alo, false, bh, (short)0, acc[t], false, false);
            acc[t] = __builtin_amdgcn_wmma_f32_16x16x32_bf16(
                false, ahi, false, bl, (short)0, acc[t], false, false);
        }

        wait_async0();        // next buffer fully landed in LDS
        __syncthreads();      // publish to all 8 waves / release cur
    }

    // C/D layout: VGPR r -> row M = mb + r + 8*(lane>>4); col N = nb+t*16+(lane&15)
    const int m0 = mb + (kg << 3);
#pragma unroll
    for (int t = 0; t < 4; ++t) {
        const int n  = nb + t * 16 + lm;
        const float bv = bias[n];
#pragma unroll
        for (int r = 0; r < 8; ++r) {
            out[(size_t)(m0 + r) * N + n] = acc[t][r] + bv;
        }
    }
}

// ---------------------------------------------------------------------------
// Block reduction helpers (wave32 shuffles + LDS across 8 waves)
// ---------------------------------------------------------------------------
__device__ __forceinline__ float waveRed(float v)
{
#pragma unroll
    for (int off = 16; off > 0; off >>= 1) v += __shfl_xor(v, off, 32);
    return v;
}

__device__ __forceinline__ float blockReduceBcast(float v, float* s8, float* res)
{
    const int lane = threadIdx.x & 31;
    const int wv   = threadIdx.x >> 5;
    v = waveRed(v);
    if (lane == 0) s8[wv] = v;
    __syncthreads();
    if (threadIdx.x == 0) {
        float t = 0.f;
#pragma unroll
        for (int i = 0; i < 8; ++i) t += s8[i];
        *res = t;
    }
    __syncthreads();
    return *res;
}

// ---------------------------------------------------------------------------
// Kernel 3: gate + value + per-group RMS norm.
// One 256-thread block per (b,t,g). Writes value -> d_out, normed -> ws.
// ---------------------------------------------------------------------------
__global__ void __launch_bounds__(256)
gate_value_kernel(const float* __restrict__ kall, const float* __restrict__ hid,
                  const float* __restrict__ vproj,
                  const float* __restrict__ wkn, const float* __restrict__ wqn,
                  const float* __restrict__ wnorm,
                  float* __restrict__ value_out, float* __restrict__ normed)
{
    __shared__ float s8[8];
    __shared__ float res[4];

    const int bid  = blockIdx.x;            // over B*T*G
    const int g    = bid & (Gg - 1);
    const int bt   = bid >> 2;              // Gg == 4
    const size_t base  = (size_t)bid * Cc;  // [B,T,G,C] flat
    const size_t vbase = (size_t)bt  * Cc;  // [B,T,C] flat
    const int tid = threadIdx.x;

    float sk = 0.f, sq = 0.f, sd = 0.f;
#pragma unroll
    for (int j = 0; j < Cc / 256; ++j) {
        const int c = tid + j * 256;
        const float kv = kall[base + c];
        const float qv = hid [base + c];
        sk = fmaf(kv, kv, sk);
        sq = fmaf(qv, qv, sq);
        sd = fmaf(kv * wkn[g * Cc + c], qv * wqn[g * Cc + c], sd);
    }
    const float skT = blockReduceBcast(sk, s8, &res[0]);
    const float sqT = blockReduceBcast(sq, s8, &res[1]);
    const float sdT = blockReduceBcast(sd, s8, &res[2]);

    const float invC = 1.0f / (float)Cc;
    const float rk = sqrtf(skT * invC + 1e-5f);
    const float rq = sqrtf(sqT * invC + 1e-5f);
    const float graw = sdT / (rk * rq) * rsqrtf((float)Cc);
    const float sgn  = (graw >= 0.f) ? 1.f : -1.f;
    const float gate = 1.f / (1.f + expf(-sgn * sqrtf(fmaxf(fabsf(graw), 1e-6f))));

    float vloc[Cc / 256];
    float sv = 0.f;
#pragma unroll
    for (int j = 0; j < Cc / 256; ++j) {
        const int c = tid + j * 256;
        const float v = vproj[vbase + c] * gate;
        vloc[j] = v;
        value_out[base + c] = v;            // residual term lives in d_out
        sv = fmaf(v, v, sv);
    }
    const float svT = blockReduceBcast(sv, s8, &res[3]);
    const float inv_rms = rsqrtf(svT * invC + 1e-5f);
#pragma unroll
    for (int j = 0; j < Cc / 256; ++j) {
        const int c = tid + j * 256;
        normed[base + c] = vloc[j] * wnorm[g * Cc + c] * inv_rms;
    }
}

// ---------------------------------------------------------------------------
// Kernel 4: dilated (DIL=3) depthwise causal conv (K=4) + SiLU + residual.
// out[idx] += silu( sum_k normed[t-(3-k)*3] * w[g,c,k] )
// ---------------------------------------------------------------------------
__global__ void __launch_bounds__(256)
conv_silu_residual(const float* __restrict__ normed, const float* __restrict__ cw,
                   float* __restrict__ out)
{
    const size_t idx = (size_t)blockIdx.x * blockDim.x + threadIdx.x;
    const int c = (int)(idx & (Cc - 1));
    size_t r = idx >> 10;                   // / Cc
    const int g = (int)(r & (Gg - 1));
    r >>= 2;                                // / Gg
    const int t = (int)(r & (Tt - 1));
    const int b = (int)(r >> 12);           // / Tt

    const float* w = cw + ((size_t)g * Cc + c) * 4;
    float acc = 0.f;
#pragma unroll
    for (int k = 0; k < 4; ++k) {
        const int tt = t - (3 - k) * 3;
        if (tt >= 0)
            acc = fmaf(normed[(((size_t)b * Tt + tt) * Gg + g) * Cc + c], w[k], acc);
    }
    const float silu = acc / (1.f + expf(-acc));
    out[idx] = out[idx] + silu;
}

// ---------------------------------------------------------------------------
// Host-side orchestration. Workspace requirement ~357 MB:
//   A hi/lo 33.5MB + Wk hi/lo 16.8MB + Wv hi/lo 4.2MB
//   + k_all 134MB + vproj 33.5MB + normed 134MB
// ---------------------------------------------------------------------------
extern "C" void kernel_launch(void* const* d_in, const int* in_sizes, int n_in,
                              void* d_out, int out_size, void* d_ws, size_t ws_size,
                              hipStream_t stream)
{
    const float* embeddings   = (const float*)d_in[0];
    const float* hidden       = (const float*)d_in[1];
    const float* key_w        = (const float*)d_in[2];
    const float* key_b        = (const float*)d_in[3];
    const float* value_w      = (const float*)d_in[4];
    const float* value_b      = (const float*)d_in[5];
    const float* w_key_norm   = (const float*)d_in[6];
    const float* w_query_norm = (const float*)d_in[7];
    const float* w_norm       = (const float*)d_in[8];
    const float* conv_w       = (const float*)d_in[9];
    float* out = (float*)d_out;

    char* ws = (char*)d_ws;
    size_t off = 0;
    auto carve = [&](size_t bytes) -> void* {
        void* p = ws + off;
        off = (off + bytes + 255) & ~(size_t)255;
        return p;
    };
    __bf16* Ahi  = (__bf16*)carve((size_t)Mm * Ee * 2);
    __bf16* Alo  = (__bf16*)carve((size_t)Mm * Ee * 2);
    __bf16* Wkhi = (__bf16*)carve((size_t)Nk * Ee * 2);
    __bf16* Wklo = (__bf16*)carve((size_t)Nk * Ee * 2);
    __bf16* Wvhi = (__bf16*)carve((size_t)Nv * Ee * 2);
    __bf16* Wvlo = (__bf16*)carve((size_t)Nv * Ee * 2);
    float*  kall   = (float*)carve((size_t)Mm * Nk * 4);
    float*  vproj  = (float*)carve((size_t)Mm * Nv * 4);
    float*  normed = (float*)carve((size_t)Bb * Tt * Gg * Cc * 4);
    (void)ws_size; (void)in_sizes; (void)n_in; (void)out_size;

    // 1) fp32 -> bf16 hi/lo splits
    split_bf16_kernel<<<2048, 256, 0, stream>>>(embeddings, Ahi, Alo, (long)Mm * Ee);
    split_bf16_kernel<<<1024, 256, 0, stream>>>(key_w,  Wkhi, Wklo, (long)Nk * Ee);
    split_bf16_kernel<<<512,  256, 0, stream>>>(value_w, Wvhi, Wvlo, (long)Nv * Ee);

    // 2) WMMA GEMMs (bf16x3, f32 accumulate): key projection and value projection
    gemm_bf16x3_wmma<<<dim3(Nk / 64, Mm / 128), 256, 0, stream>>>(
        Ahi, Alo, Wkhi, Wklo, key_b, kall, Nk, Ee);
    gemm_bf16x3_wmma<<<dim3(Nv / 64, Mm / 128), 256, 0, stream>>>(
        Ahi, Alo, Wvhi, Wvlo, value_b, vproj, Nv, Ee);

    // 3) gate + value (-> d_out) + per-group RMS norm (-> ws.normed)
    gate_value_kernel<<<Bb * Tt * Gg, 256, 0, stream>>>(
        kall, hidden, vproj, w_key_norm, w_query_norm, w_norm, out, normed);

    // 4) dilated depthwise conv + SiLU + residual (in-place on d_out)
    conv_silu_residual<<<(Bb * Tt * Gg * Cc) / 256, 256, 0, stream>>>(
        normed, conv_w, out);
}
